// VQLayer_18116172054711
// MI455X (gfx1250) — matrix operand order repaired and checked by reading, not compiled, not run on hardware
//
#include <hip/hip_runtime.h>
#include <hip/hip_bf16.h>
#include <math.h>

// ---------------------------------------------------------------------------
// VQ-VAE layer on MI455X (gfx1250): bf16 WMMA (f32 accum) for all GEMMs.
// LDS tiles are stored in WMMA *fragment order* so fragment loads are single
// 256-bit ds reads and staging stores are 64-bit ds writes. The codebook is
// pre-converted to bf16 once so the distance-GEMM inner loop has no cvt ops.
// ---------------------------------------------------------------------------

typedef __attribute__((ext_vector_type(16))) __bf16 v16bf;
typedef __attribute__((ext_vector_type(4)))  __bf16 v4bf;
typedef __attribute__((ext_vector_type(8)))  float  v8f;

#define ENC_DIM 1024
#define EMB     256
#define KCODES  8192
#define ROWS    8192                 // 4 * 2048
#define NELEM   (ROWS * ENC_DIM)     // 8388608
#define NVEC    (ROWS * 4)           // 32768 flat vectors of dim 256
#define COMMIT  0.25f

static __device__ __forceinline__ v8f wmma_bf16(v16bf a, v16bf b, v8f c) {
  // D = A(16x32 bf16) * B(32x16 bf16) + C(16x16 f32)
  return __builtin_amdgcn_wmma_f32_16x16x32_bf16(false, a, false, b,
                                                 (short)0, c, false, false);
}

// Fragment-order mapping used here for the 16-bit 16x32 A matrix:
//   element (row r in tile, k c in 0..31) lives in
//   lane   = (r & 15) + 16 * (c >> 4)
//   half   = c & 15                           (K-contiguous within a lane)
// Fragment-order mapping for the 16-bit 32x16 B matrix:
//   element (k, col n in tile) lives in
//   lane   = (n & 15) + 16 * (k >> 4)
//   half   = k & 15                           (K-contiguous within a lane)

// ---------------------------------------------------------------------------
// GEMM: C[8192,1024] = relu(A[8192,1024] @ W[1024,1024] + bias)
// Block tile 128x128, 8 waves in 4x2, wave tile 32x64 (8 wmma / step), BK=32.
// ---------------------------------------------------------------------------
#define BM 128
#define BN 128
#define BK 32

__global__ __launch_bounds__(256)
void gemm_bias_relu(const float* __restrict__ A, const float* __restrict__ W,
                    const float* __restrict__ bias, float* __restrict__ C) {
  __shared__ v16bf sAf[8][32];   // 8 m-tiles x 32 lanes x 32 B = 8 KB
  __shared__ v16bf sBf[8][32];   // 8 n-tiles x 32 lanes x 32 B = 8 KB

  const int tid  = threadIdx.x;
  const int wid  = tid >> 5;
  const int lane = tid & 31;
  const int hi   = lane >> 4;

  const int bm = blockIdx.x * BM;
  const int bn = blockIdx.y * BN;

  const int wave_m = wid & 3;    // 0..3 -> 32 rows each
  const int wave_n = wid >> 2;   // 0..1 -> 64 cols each
  const int mt0 = wave_m * 2, mt1 = mt0 + 1;
  const int ntb = wave_n * 4;

  v8f acc0[4] = {}, acc1[4] = {};

  for (int k0 = 0; k0 < ENC_DIM; k0 += BK) {
    // ---- Stage A tile (128x32): 1024 float4 units, 4 per thread ----
    #pragma unroll
    for (int s = 0; s < 4; s++) {
      int u  = tid + 256 * s;
      int r  = u >> 3;                 // 0..127
      int c0 = (u & 7) << 2;           // 0,4,...,28
      float4 a4 = *(const float4*)&A[(size_t)(bm + r) * ENC_DIM + k0 + c0];
      v4bf q;
      q[0] = (__bf16)a4.x; q[1] = (__bf16)a4.y;
      q[2] = (__bf16)a4.z; q[3] = (__bf16)a4.w;
      int mt = r >> 4;
      int ln = (r & 15) + ((c0 >> 4) << 4);
      int hb = c0 & 15;
      *(v4bf*)((__bf16*)&sAf[mt][ln] + hb) = q;   // ds_store_b64
    }
    // ---- Stage W tile (32x128): 1024 units of 4-k-deep, 4 per thread ----
    #pragma unroll
    for (int s = 0; s < 4; s++) {
      int u   = tid + 256 * s;
      int kq  = u >> 7;                // 0..7
      int col = u & 127;               // coalesced across threads
      const float* wp = &W[(size_t)(k0 + kq * 4) * ENC_DIM + bn + col];
      v4bf q;
      q[0] = (__bf16)wp[0];
      q[1] = (__bf16)wp[ENC_DIM];
      q[2] = (__bf16)wp[2 * ENC_DIM];
      q[3] = (__bf16)wp[3 * ENC_DIM];
      int nt = col >> 4;
      int ln = (col & 15) + ((kq >> 2) << 4);
      int hb = (kq & 3) << 2;
      *(v4bf*)((__bf16*)&sBf[nt][ln] + hb) = q;   // ds_store_b64
    }
    __syncthreads();

    if (k0 + BK < ENC_DIM) {   // global_prefetch_b8 of the next K tiles
      __builtin_prefetch(&A[(size_t)(bm + (tid & 127)) * ENC_DIM + k0 + BK], 0, 1);
      __builtin_prefetch(&W[(size_t)(k0 + BK + (tid & 31)) * ENC_DIM + bn], 0, 1);
    }

    // ---- Fragment loads: single 256-bit LDS reads ----
    v16bf af0 = sAf[mt0][lane];
    v16bf af1 = sAf[mt1][lane];
    #pragma unroll
    for (int t = 0; t < 4; t++) {
      v16bf bf = sBf[ntb + t][lane];
      acc0[t] = wmma_bf16(af0, bf, acc0[t]);
      acc1[t] = wmma_bf16(af1, bf, acc1[t]);
    }
    __syncthreads();
  }

  // Epilogue: bias + ReLU.  C layout: n = lane&15, m = v + 8*(lane>>4)
  #pragma unroll
  for (int t = 0; t < 4; t++) {
    const int col = bn + (ntb + t) * 16 + (lane & 15);
    const float bv = bias[col];
    #pragma unroll
    for (int v = 0; v < 8; v++) {
      const int r0 = bm + mt0 * 16 + v + (hi << 3);
      const int r1 = bm + mt1 * 16 + v + (hi << 3);
      float x;
      x = acc0[t][v] + bv; C[(size_t)r0 * ENC_DIM + col] = x > 0.f ? x : 0.f;
      x = acc1[t][v] + bv; C[(size_t)r1 * ENC_DIM + col] = x > 0.f ? x : 0.f;
    }
  }
}

// ---------------------------------------------------------------------------
// LayerNorm over rows of 1024 (one float4 per thread).
// ---------------------------------------------------------------------------
__global__ __launch_bounds__(256)
void layernorm_k(const float* __restrict__ X, const float* __restrict__ g,
                 const float* __restrict__ b, float* __restrict__ Y) {
  const int tid = threadIdx.x, wid = tid >> 5, lane = tid & 31;
  const size_t row = blockIdx.x;
  const float4* x4 = (const float4*)(X + row * ENC_DIM);

  float4 v = x4[tid];
  float s  = v.x + v.y + v.z + v.w;
  float ss = v.x * v.x + v.y * v.y + v.z * v.z + v.w * v.w;
  #pragma unroll
  for (int off = 16; off > 0; off >>= 1) {
    s  += __shfl_xor(s, off, 32);
    ss += __shfl_xor(ss, off, 32);
  }
  __shared__ float rs[8], rss[8];
  __shared__ float s_mean, s_inv;
  if (lane == 0) { rs[wid] = s; rss[wid] = ss; }
  __syncthreads();
  if (tid == 0) {
    float S = 0.f, SS = 0.f;
    #pragma unroll
    for (int w = 0; w < 8; w++) { S += rs[w]; SS += rss[w]; }
    float mean = S * (1.f / ENC_DIM);
    float var  = SS * (1.f / ENC_DIM) - mean * mean;
    s_mean = mean;
    s_inv  = rsqrtf(var + 1e-5f);
  }
  __syncthreads();
  const float mean = s_mean, inv = s_inv;
  float4 g4 = ((const float4*)g)[tid];
  float4 b4 = ((const float4*)b)[tid];
  float4 o;
  o.x = (v.x - mean) * inv * g4.x + b4.x;
  o.y = (v.y - mean) * inv * g4.y + b4.y;
  o.z = (v.z - mean) * inv * g4.z + b4.z;
  o.w = (v.w - mean) * inv * g4.w + b4.w;
  ((float4*)(Y + row * ENC_DIM))[tid] = o;
}

// ---------------------------------------------------------------------------
// Codebook squared norms + one-time fp32 -> bf16 codebook conversion.
// ---------------------------------------------------------------------------
__global__ __launch_bounds__(256)
void cnorm_k(const float* __restrict__ cb, float* __restrict__ cnorm,
             __bf16* __restrict__ cbh) {
  const int tid = threadIdx.x, wid = tid >> 5, lane = tid & 31;
  const int row = blockIdx.x * 8 + wid;
  const float4* r4 = (const float4*)(cb + (size_t)row * EMB);
  float s = 0.f;
  #pragma unroll
  for (int c = lane; c < EMB / 4; c += 32) {
    float4 v = r4[c];
    s += v.x * v.x + v.y * v.y + v.z * v.z + v.w * v.w;
    v4bf q;
    q[0] = (__bf16)v.x; q[1] = (__bf16)v.y;
    q[2] = (__bf16)v.z; q[3] = (__bf16)v.w;
    *(v4bf*)&cbh[(size_t)row * EMB + c * 4] = q;   // global_store_b64
  }
  #pragma unroll
  for (int off = 16; off > 0; off >>= 1) s += __shfl_xor(s, off, 32);
  if (lane == 0) cnorm[row] = s;
}

// ---------------------------------------------------------------------------
// VQ argmin: 16 z-rows per block staged in fragment order; each wave scans a
// disjoint 1024-code slice with 8 bf16 WMMAs per 16-code tile (K = 256).
// B fragments load directly from the bf16 codebook (K-contiguous halves).
// score = ||c||^2 - 2 z.c  (||z||^2 constant per row, irrelevant to argmin)
// ---------------------------------------------------------------------------
__global__ __launch_bounds__(256)
void vq_argmin(const float* __restrict__ z, const __bf16* __restrict__ cbh,
               const float* __restrict__ cnorm, int* __restrict__ idx) {
  __shared__ v16bf sZf[8][32];       // 8 k-chunks x 32 lanes x 32 B = 8 KB
  __shared__ float redv[8][16];
  __shared__ int   redi[8][16];

  const int tid = threadIdx.x, wid = tid >> 5, lane = tid & 31;
  const int hi = lane >> 4;
  const int row0 = blockIdx.x * 16;

  // Stage 16x256 z rows -> fragment order (1024 float4 units, 4 per thread)
  #pragma unroll
  for (int s = 0; s < 4; s++) {
    int u  = tid + 256 * s;
    int r  = u >> 6;                  // 0..15
    int c0 = (u & 63) << 2;           // 0..252 step 4
    float4 a4 = *(const float4*)&z[(size_t)(row0 + r) * EMB + c0];
    v4bf q;
    q[0] = (__bf16)a4.x; q[1] = (__bf16)a4.y;
    q[2] = (__bf16)a4.z; q[3] = (__bf16)a4.w;
    int kc = c0 >> 5;
    int cc = c0 & 31;
    int ln = r + ((cc >> 4) << 4);
    int hb = cc & 15;
    *(v4bf*)((__bf16*)&sZf[kc][ln] + hb) = q;
  }
  __syncthreads();

  v16bf afr[8];
  #pragma unroll
  for (int kc = 0; kc < 8; kc++) afr[kc] = sZf[kc][lane];

  float best[8];
  int   bidx[8];
  #pragma unroll
  for (int v = 0; v < 8; v++) { best[v] = 3.4e38f; bidx[v] = 0; }

  const int nbase = wid * (KCODES / 8);      // 1024 codes per wave
  for (int n0 = nbase; n0 < nbase + KCODES / 8; n0 += 16) {
    const int code = n0 + (lane & 15);
    if (n0 + 16 < nbase + KCODES / 8)
      __builtin_prefetch(&cbh[(size_t)(code + 16) * EMB + (hi << 4)], 0, 1);

    const __bf16* crow = cbh + (size_t)code * EMB + (hi << 4);
    v8f acc = {};
    #pragma unroll
    for (int kc = 0; kc < 8; kc++) {
      // B fragment halves are K-ordered: half t <-> k = kc*32 + hi*16 + t,
      // i.e. 32 contiguous bytes of the bf16 codebook row.
      v16bf bfr = *(const v16bf*)(crow + kc * 32);
      acc = wmma_bf16(afr[kc], bfr, acc);
    }
    const float cn = cnorm[code];
    #pragma unroll
    for (int v = 0; v < 8; v++) {
      float sc = cn - 2.f * acc[v];
      if (sc < best[v]) { best[v] = sc; bidx[v] = code; }
    }
  }

  // Reduce across the 16 lanes sharing the same m set (C-layout striping).
  #pragma unroll
  for (int off = 8; off > 0; off >>= 1) {
    #pragma unroll
    for (int v = 0; v < 8; v++) {
      float ov = __shfl_xor(best[v], off, 16);
      int   oi = __shfl_xor(bidx[v], off, 16);
      if (ov < best[v] || (ov == best[v] && oi < bidx[v])) {
        best[v] = ov; bidx[v] = oi;
      }
    }
  }
  if ((lane & 15) == 0) {
    #pragma unroll
    for (int v = 0; v < 8; v++) {
      redv[wid][(hi << 3) + v] = best[v];
      redi[wid][(hi << 3) + v] = bidx[v];
    }
  }
  __syncthreads();
  if (tid < 16) {
    float bv = redv[0][tid]; int bi = redi[0][tid];
    #pragma unroll
    for (int w = 1; w < 8; w++) {
      float v = redv[w][tid]; int i = redi[w][tid];
      if (v < bv || (v == bv && i < bi)) { bv = v; bi = i; }
    }
    idx[row0 + tid] = bi;
  }
}

// ---------------------------------------------------------------------------
// Code-usage histogram (deterministic integer atomics).
// ---------------------------------------------------------------------------
__global__ __launch_bounds__(256)
void vq_hist(const int* __restrict__ idx, int* __restrict__ counts) {
  int r = blockIdx.x * 256 + threadIdx.x;   // exactly NVEC threads
  atomicAdd(&counts[idx[r]], 1);
}

// ---------------------------------------------------------------------------
// Gather quantized vectors (float4) + per-block partial sum of (q - z)^2.
// ---------------------------------------------------------------------------
__global__ __launch_bounds__(256)
void vq_gather_mse(const float* __restrict__ z, const float* __restrict__ cb,
                   const int* __restrict__ idx, float* __restrict__ Q,
                   float* __restrict__ part) {
  const int tid = threadIdx.x, wid = tid >> 5, lane = tid & 31;
  const size_t e = ((size_t)blockIdx.x * 256 + tid) * 4;   // element index
  const int row = (int)(e >> 8);
  const int c   = (int)(e & 255);
  float4 q4 = *(const float4*)&cb[(size_t)idx[row] * EMB + c];
  float4 z4 = *(const float4*)&z[e];
  *(float4*)&Q[e] = q4;              // straight-through value == q
  float dx = q4.x - z4.x, dy = q4.y - z4.y, dz = q4.z - z4.z, dw = q4.w - z4.w;
  float s = dx * dx + dy * dy + dz * dz + dw * dw;

  #pragma unroll
  for (int off = 16; off > 0; off >>= 1) s += __shfl_xor(s, off, 32);
  __shared__ float rs[8];
  if (lane == 0) rs[wid] = s;
  __syncthreads();
  if (tid == 0) {
    float S = 0.f;
    #pragma unroll
    for (int w = 0; w < 8; w++) S += rs[w];
    part[blockIdx.x] = S;
  }
}

// ---------------------------------------------------------------------------
// Finalize: loss = (1 + COMMIT) * MSE ; perplexity from histogram.
// ---------------------------------------------------------------------------
__global__ __launch_bounds__(256)
void vq_finalize(const float* __restrict__ part, const int* __restrict__ counts,
                 float* __restrict__ out2) {
  const int tid = threadIdx.x, wid = tid >> 5, lane = tid & 31;
  float s = 0.f, h = 0.f;
  for (int i = tid; i < NELEM / 1024; i += 256) s += part[i];   // 8192 partials
  for (int i = tid; i < KCODES; i += 256) {
    float p = (float)counts[i] * (1.f / (float)NVEC);
    h += p * logf(p + 1e-10f);
  }
  #pragma unroll
  for (int off = 16; off > 0; off >>= 1) {
    s += __shfl_xor(s, off, 32);
    h += __shfl_xor(h, off, 32);
  }
  __shared__ float rs[8], rh[8];
  if (lane == 0) { rs[wid] = s; rh[wid] = h; }
  __syncthreads();
  if (tid == 0) {
    float S = 0.f, H = 0.f;
    #pragma unroll
    for (int w = 0; w < 8; w++) { S += rs[w]; H += rh[w]; }
    float mse = S * (1.f / (float)NELEM);
    out2[0] = (1.f + COMMIT) * mse;  // q_latent + COMMIT * e_latent (same value)
    out2[1] = expf(-H);              // perplexity
  }
}

// ---------------------------------------------------------------------------
// Launcher
// ---------------------------------------------------------------------------
extern "C" void kernel_launch(void* const* d_in, const int* in_sizes, int n_in,
                              void* d_out, int out_size, void* d_ws, size_t ws_size,
                              hipStream_t stream) {
  const float* x    = (const float*)d_in[0];
  const float* We1  = (const float*)d_in[1];
  const float* be1  = (const float*)d_in[2];
  const float* ge1  = (const float*)d_in[3];
  const float* bne1 = (const float*)d_in[4];
  const float* We2  = (const float*)d_in[5];
  const float* be2  = (const float*)d_in[6];
  const float* ge2  = (const float*)d_in[7];
  const float* bne2 = (const float*)d_in[8];
  const float* Wd1  = (const float*)d_in[9];
  const float* bd1  = (const float*)d_in[10];
  const float* gd1  = (const float*)d_in[11];
  const float* bnd1 = (const float*)d_in[12];
  const float* Wd2  = (const float*)d_in[13];
  const float* bd2  = (const float*)d_in[14];
  const float* gd2  = (const float*)d_in[15];
  const float* bnd2 = (const float*)d_in[16];
  const float* cb   = (const float*)d_in[17];

  float* out = (float*)d_out;

  float*  A      = (float*)d_ws;              // 8388608 f32
  float*  B      = A + NELEM;                 // 8388608 f32
  float*  cnorm  = B + NELEM;                 // 8192 f32
  int*    idxb   = (int*)(cnorm + KCODES);    // 32768 i32
  int*    counts = idxb + NVEC;               // 8192 i32
  float*  part   = (float*)(counts + KCODES); // 8192 f32
  __bf16* cbh    = (__bf16*)(part + NELEM / 1024); // 8192*256 bf16 (4 MB)

  hipMemsetAsync(counts, 0, KCODES * sizeof(int), stream);

  dim3 ggrid(ROWS / BM, ENC_DIM / BN);        // (64, 8)

  cnorm_k<<<KCODES / 8, 256, 0, stream>>>(cb, cnorm, cbh);

  // Encoder
  gemm_bias_relu<<<ggrid, 256, 0, stream>>>(x, We1, be1, A);
  layernorm_k<<<ROWS, 256, 0, stream>>>(A, ge1, bne1, B);
  gemm_bias_relu<<<ggrid, 256, 0, stream>>>(B, We2, be2, A);
  layernorm_k<<<ROWS, 256, 0, stream>>>(A, ge2, bne2, B);   // B = z

  // Vector quantizer
  vq_argmin<<<NVEC / 16, 256, 0, stream>>>(B, cbh, cnorm, idxb);
  vq_hist<<<NVEC / 256, 256, 0, stream>>>(idxb, counts);
  vq_gather_mse<<<NELEM / 1024, 256, 0, stream>>>(B, cb, idxb, A, part); // A = q
  vq_finalize<<<1, 256, 0, stream>>>(part, counts, out + NELEM);

  // Decoder
  gemm_bias_relu<<<ggrid, 256, 0, stream>>>(A, Wd1, bd1, B);
  layernorm_k<<<ROWS, 256, 0, stream>>>(B, gd1, bnd1, A);
  gemm_bias_relu<<<ggrid, 256, 0, stream>>>(A, Wd2, bd2, B);
  layernorm_k<<<ROWS, 256, 0, stream>>>(B, gd2, bnd2, out);
}
